// MixtralMoE_41832981463129
// MI455X (gfx1250) — matrix operand order repaired and checked by reading, not compiled, not run on hardware
//
#include <hip/hip_runtime.h>

// ---------------- problem constants (from reference setup_inputs) -----------
#define TDIM 2048   // tokens
#define HDIM 2048   // hidden
#define IDIM 7168   // intermediate
#define EXP  8      // experts
#define TOPK 2
#define ROWS_MAX (TDIM * TOPK)   // 4096 routed rows total

#define SP 40            // LDS row pitch in bf16 elems (32 data + 8 pad -> 80B)
#define ASZ (64 * SP)    // one A buffer (bf16 elems)
#define BSZ (128 * SP)   // one B buffer (bf16 elems)

typedef unsigned short u16;
typedef __attribute__((ext_vector_type(16))) __bf16 v16bf;
typedef __attribute__((ext_vector_type(8)))  float  v8f;

// ---------------- helpers ---------------------------------------------------
__device__ __forceinline__ u16 f2bf_u(float f) {
  unsigned int u = __float_as_uint(f);
  u += 0x7FFFu + ((u >> 16) & 1u);        // round-to-nearest-even
  return (u16)(u >> 16);
}

__device__ __forceinline__ uint4 pack8(float4 a, float4 b) {
  uint4 r;
  r.x = (unsigned)f2bf_u(a.x) | ((unsigned)f2bf_u(a.y) << 16);
  r.y = (unsigned)f2bf_u(a.z) | ((unsigned)f2bf_u(a.w) << 16);
  r.z = (unsigned)f2bf_u(b.x) | ((unsigned)f2bf_u(b.y) << 16);
  r.w = (unsigned)f2bf_u(b.z) | ((unsigned)f2bf_u(b.w) << 16);
  return r;
}

union Frag16 { uint4 q[2]; v16bf v; };

// A operand, 16x32 bf16 (ISA 7.12.2): lane<16 -> row=lane,   K = {0..7, 16..23}
//                                     lane>=16 -> row=lane-16,K = {8..15,24..31}
__device__ __forceinline__ v16bf ld_fragA(const u16* base, int lane) {
  const u16* p = base + (lane & 15) * SP + (lane >> 4) * 8;
  Frag16 f;
  f.q[0] = *(const uint4*)(p);
  f.q[1] = *(const uint4*)(p + 16);
  return f.v;
}

// B operand, 32x16 bf16: lane&15 -> col n; lanes 0-15 hold K={0..7,16..23},
// lanes 16-31 hold K={8..15,24..31} (same order as A)
__device__ __forceinline__ v16bf ld_fragB(const u16* base, int lane) {
  const u16* p = base + (lane & 15) * SP + (lane >> 4) * 8;
  Frag16 f;
  f.q[0] = *(const uint4*)(p);
  f.q[1] = *(const uint4*)(p + 16);
  return f.v;
}

__device__ __forceinline__ v8f wmma_bf16(v16bf a, v16bf b, v8f c) {
  return __builtin_amdgcn_wmma_f32_16x16x32_bf16(false, a, false, b, (short)0, c, false, false);
}

// convert 16 consecutive fp32 -> 16 bf16 into LDS
__device__ __forceinline__ void cvt16_to_lds(u16* dst, const float* src) {
  const float4* s = (const float4*)src;
  float4 f0 = s[0], f1 = s[1], f2 = s[2], f3 = s[3];
  ((uint4*)dst)[0] = pack8(f0, f1);
  ((uint4*)dst)[1] = pack8(f2, f3);
}

// fast sigmoid: v_rcp_f32 instead of the IEEE division chain
__device__ __forceinline__ float sigmoid_fast(float x) {
  return __builtin_amdgcn_rcpf(1.f + __expf(-x));
}

// ---------------- kernel 0: zero output + counters --------------------------
__global__ void k_zero(float* __restrict__ out, int* __restrict__ counts, long n) {
  long i = (long)blockIdx.x * blockDim.x + threadIdx.x;
  long step = (long)gridDim.x * blockDim.x;
  for (; i < n; i += step) out[i] = 0.f;
  if (blockIdx.x == 0 && threadIdx.x < EXP) counts[threadIdx.x] = 0;
}

// ---------------- kernel 1: router (logits -> top2 -> renorm weights) -------
__global__ void k_router(const float* __restrict__ hidden, const float* __restrict__ gate,
                         int* __restrict__ tok_e, float* __restrict__ tok_w,
                         int* __restrict__ counts) {
  int t = blockIdx.x * blockDim.x + threadIdx.x;
  if (t >= TDIM) return;
  const float4* hp = (const float4*)(hidden + (size_t)t * HDIM);
  float l[EXP];
#pragma unroll
  for (int e = 0; e < EXP; ++e) l[e] = 0.f;
  for (int k = 0; k < HDIM / 4; ++k) {
    float4 h4 = hp[k];
#pragma unroll
    for (int e = 0; e < EXP; ++e) {
      float4 g4 = ((const float4*)(gate + (size_t)e * HDIM))[k];
      l[e] += h4.x * g4.x + h4.y * g4.y + h4.z * g4.z + h4.w * g4.w;
    }
  }
  int e0 = 0;
#pragma unroll
  for (int e = 1; e < EXP; ++e) if (l[e] > l[e0]) e0 = e;
  int e1 = (e0 == 0) ? 1 : 0;
#pragma unroll
  for (int e = 0; e < EXP; ++e) if (e != e0 && l[e] > l[e1]) e1 = e;
  // softmax denominators cancel under top-2 renormalization
  float w0 = 1.f / (1.f + __expf(l[e1] - l[e0]));
  float w1 = 1.f - w0;
  tok_e[2 * t] = e0;     tok_e[2 * t + 1] = e1;
  tok_w[2 * t] = w0;     tok_w[2 * t + 1] = w1;
  atomicAdd(&counts[e0], 1);
  atomicAdd(&counts[e1], 1);
}

// ---------------- kernel 2: exclusive scan over 8 experts -------------------
__global__ void k_scan(const int* __restrict__ counts, int* __restrict__ offsets,
                       int* __restrict__ cursors) {
  if (threadIdx.x == 0 && blockIdx.x == 0) {
    int s = 0;
    for (int e = 0; e < EXP; ++e) { offsets[e] = s; cursors[e] = s; s += counts[e]; }
  }
}

// ---------------- kernel 3: scatter tokens into per-expert row lists --------
__global__ void k_assign(const int* __restrict__ tok_e, const float* __restrict__ tok_w,
                         int* __restrict__ cursors, int* __restrict__ row_token,
                         float* __restrict__ row_weight) {
  int t = blockIdx.x * blockDim.x + threadIdx.x;
  if (t >= TDIM) return;
#pragma unroll
  for (int j = 0; j < TOPK; ++j) {
    int e = tok_e[2 * t + j];
    int pos = atomicAdd(&cursors[e], 1);
    row_token[pos] = t;
    row_weight[pos] = tok_w[2 * t + j];
  }
}

// ---------------- kernel 4: GEMM1 — act = silu(X@w1^T) * (X@w3^T) -----------
// block tile 64(M) x 128(N), 8 waves of 32x32, K-step 32, bf16 WMMA,
// double-buffered LDS (1 barrier per K-step), prefetch of next weight tile
__global__ __launch_bounds__(256) void k_gemm1(
    const float* __restrict__ hidden, const float* __restrict__ w1,
    const float* __restrict__ w3, const int* __restrict__ row_token,
    const int* __restrict__ offsets, const int* __restrict__ counts,
    u16* __restrict__ act) {
  int e = blockIdx.z;
  int cnt = counts[e];
  int m_base = blockIdx.x * 64;
  if (m_base >= cnt) return;
  int row0 = offsets[e];
  int n_base = blockIdx.y * 128;
  const float* W1 = w1 + (size_t)e * IDIM * HDIM;
  const float* W3 = w3 + (size_t)e * IDIM * HDIM;

  __shared__ __attribute__((aligned(16))) u16 As[2 * ASZ];
  __shared__ __attribute__((aligned(16))) u16 Bs1[2 * BSZ];
  __shared__ __attribute__((aligned(16))) u16 Bs3[2 * BSZ];

  int tid = threadIdx.x;
  int lane = tid & 31, wv = tid >> 5;
  int wm = wv & 1, wn = wv >> 1;                 // 2 x 4 wave grid

  int ar = tid >> 2, akg = tid & 3;              // A fill: 64 rows x 4 chunks of 8
  int arow = m_base + ar;
  int atok = (arow < cnt) ? row_token[row0 + arow] : -1;
  int bn = tid >> 1, bkg = tid & 1;              // B fill: 128 rows x 2 chunks of 16

  v8f a00 = {}, a01 = {}, a10 = {}, a11 = {};    // w1 accumulators
  v8f c00 = {}, c01 = {}, c10 = {}, c11 = {};    // w3 accumulators

  const float* b1row = W1 + (size_t)(n_base + bn) * HDIM;
  const float* b3row = W3 + (size_t)(n_base + bn) * HDIM;

  auto FILL = [&](int k0, int b) {
    float4 fz = {0.f, 0.f, 0.f, 0.f};
    float4 f0 = fz, f1 = fz;
    if (atok >= 0) {
      const float4* s = (const float4*)(hidden + (size_t)atok * HDIM + k0 + akg * 8);
      f0 = s[0]; f1 = s[1];
    }
    *(uint4*)(As + b * ASZ + ar * SP + akg * 8) = pack8(f0, f1);
    cvt16_to_lds(Bs1 + b * BSZ + bn * SP + bkg * 16, b1row + k0 + bkg * 16);
    cvt16_to_lds(Bs3 + b * BSZ + bn * SP + bkg * 16, b3row + k0 + bkg * 16);
    // pull the next K-tile of weights toward the caches while this one computes
    __builtin_prefetch(b1row + k0 + 32 + bkg * 16, 0, 0);
    __builtin_prefetch(b3row + k0 + 32 + bkg * 16, 0, 0);
  };

  auto COMP = [&](int b) {
    const u16* A = As + b * ASZ;
    const u16* B1 = Bs1 + b * BSZ;
    const u16* B3 = Bs3 + b * BSZ;
    v16bf fa0 = ld_fragA(A + (wm * 32) * SP, lane);
    v16bf fa1 = ld_fragA(A + (wm * 32 + 16) * SP, lane);
    v16bf b10 = ld_fragB(B1 + (wn * 32) * SP, lane);
    v16bf b11 = ld_fragB(B1 + (wn * 32 + 16) * SP, lane);
    v16bf b30 = ld_fragB(B3 + (wn * 32) * SP, lane);
    v16bf b31 = ld_fragB(B3 + (wn * 32 + 16) * SP, lane);
    a00 = wmma_bf16(fa0, b10, a00);  a01 = wmma_bf16(fa0, b11, a01);
    a10 = wmma_bf16(fa1, b10, a10);  a11 = wmma_bf16(fa1, b11, a11);
    c00 = wmma_bf16(fa0, b30, c00);  c01 = wmma_bf16(fa0, b31, c01);
    c10 = wmma_bf16(fa1, b30, c10);  c11 = wmma_bf16(fa1, b31, c11);
  };

  FILL(0, 0);
  __syncthreads();
  int buf = 0;
  for (int k0 = 32; k0 < HDIM; k0 += 32) {
    FILL(k0, buf ^ 1);   // overlaps with compute on `buf`
    COMP(buf);
    __syncthreads();
    buf ^= 1;
  }
  COMP(buf);

  // epilogue: C layout — lane<16: M=v, N=lane; lane>=16: M=v+8, N=lane-16
  int nloc = lane & 15;
  int mb = (lane >> 4) * 8;
#pragma unroll
  for (int pq = 0; pq < 4; ++pq) {
    int P = pq >> 1, Q = pq & 1;
    v8f vg = (pq == 0) ? a00 : (pq == 1) ? a01 : (pq == 2) ? a10 : a11;
    v8f vu = (pq == 0) ? c00 : (pq == 1) ? c01 : (pq == 2) ? c10 : c11;
    int rbase = m_base + wm * 32 + P * 16 + mb;
    int col = n_base + wn * 32 + Q * 16 + nloc;
#pragma unroll
    for (int v = 0; v < 8; ++v) {
      int rowg = rbase + v;
      if (rowg < cnt) {
        float x = vg[v], y = vu[v];
        float g = x * sigmoid_fast(x) * y;        // silu(x) * y
        act[(size_t)(row0 + rowg) * IDIM + col] = f2bf_u(g);
      }
    }
  }
}

// ---------------- kernel 5: GEMM2 — out += w * (act @ w2^T) -----------------
__global__ __launch_bounds__(256) void k_gemm2(
    const u16* __restrict__ act, const float* __restrict__ w2,
    const int* __restrict__ row_token, const float* __restrict__ row_weight,
    const int* __restrict__ offsets, const int* __restrict__ counts,
    float* __restrict__ out) {
  int e = blockIdx.z;
  int cnt = counts[e];
  int m_base = blockIdx.x * 64;
  if (m_base >= cnt) return;
  int row0 = offsets[e];
  int n_base = blockIdx.y * 128;
  const float* W2 = w2 + (size_t)e * HDIM * IDIM;

  __shared__ __attribute__((aligned(16))) u16 As[2 * ASZ];
  __shared__ __attribute__((aligned(16))) u16 Bs[2 * BSZ];

  int tid = threadIdx.x;
  int lane = tid & 31, wv = tid >> 5;
  int wm = wv & 1, wn = wv >> 1;

  int ar = tid >> 2, akg = tid & 3;
  int arow = m_base + ar;
  int bn = tid >> 1, bkg = tid & 1;

  const u16* arowp = (arow < cnt) ? (act + (size_t)(row0 + arow) * IDIM) : (const u16*)0;
  const float* brow = W2 + (size_t)(n_base + bn) * IDIM;

  v8f a00 = {}, a01 = {}, a10 = {}, a11 = {};

  auto FILL = [&](int k0, int b) {
    uint4 d = {0u, 0u, 0u, 0u};
    if (arowp) d = *(const uint4*)(arowp + k0 + akg * 8);
    *(uint4*)(As + b * ASZ + ar * SP + akg * 8) = d;
    cvt16_to_lds(Bs + b * BSZ + bn * SP + bkg * 16, brow + k0 + bkg * 16);
    __builtin_prefetch(brow + k0 + 32 + bkg * 16, 0, 0);
  };

  auto COMP = [&](int b) {
    const u16* A = As + b * ASZ;
    const u16* B = Bs + b * BSZ;
    v16bf fa0 = ld_fragA(A + (wm * 32) * SP, lane);
    v16bf fa1 = ld_fragA(A + (wm * 32 + 16) * SP, lane);
    v16bf fb0 = ld_fragB(B + (wn * 32) * SP, lane);
    v16bf fb1 = ld_fragB(B + (wn * 32 + 16) * SP, lane);
    a00 = wmma_bf16(fa0, fb0, a00);  a01 = wmma_bf16(fa0, fb1, a01);
    a10 = wmma_bf16(fa1, fb0, a10);  a11 = wmma_bf16(fa1, fb1, a11);
  };

  FILL(0, 0);
  __syncthreads();
  int buf = 0;
  for (int k0 = 32; k0 < IDIM; k0 += 32) {
    FILL(k0, buf ^ 1);
    COMP(buf);
    __syncthreads();
    buf ^= 1;
  }
  COMP(buf);

  int nloc = lane & 15;
  int mb = (lane >> 4) * 8;
#pragma unroll
  for (int P = 0; P < 2; ++P) {
    v8f v0 = (P == 0) ? a00 : a10;
    v8f v1 = (P == 0) ? a01 : a11;
    int rbase = m_base + wm * 32 + P * 16 + mb;
    int col0 = n_base + wn * 32 + nloc;
#pragma unroll
    for (int v = 0; v < 8; ++v) {
      int rowg = rbase + v;
      if (rowg < cnt) {
        int tok = row_token[row0 + rowg];
        float wgt = row_weight[row0 + rowg];
        // exactly 2 contributions per output element -> order-independent sum
        atomicAdd(out + (size_t)tok * HDIM + col0, wgt * v0[v]);
        atomicAdd(out + (size_t)tok * HDIM + col0 + 16, wgt * v1[v]);
      }
    }
  }
}

// ---------------- host-side launcher ----------------------------------------
extern "C" void kernel_launch(void* const* d_in, const int* in_sizes, int n_in,
                              void* d_out, int out_size, void* d_ws, size_t ws_size,
                              hipStream_t stream) {
  (void)in_sizes; (void)n_in; (void)out_size; (void)ws_size;
  const float* hidden = (const float*)d_in[0];
  const float* gate_w = (const float*)d_in[1];
  const float* w1     = (const float*)d_in[2];
  const float* w3     = (const float*)d_in[3];
  const float* w2     = (const float*)d_in[4];
  float* out = (float*)d_out;

  // workspace layout
  char* ws = (char*)d_ws;
  size_t off = 0;
  u16*   act        = (u16*)(ws + off);  off += (size_t)ROWS_MAX * IDIM * sizeof(u16);   // 58.7 MB
  int*   row_token  = (int*)(ws + off);  off += (size_t)ROWS_MAX * sizeof(int);
  float* row_weight = (float*)(ws + off); off += (size_t)ROWS_MAX * sizeof(float);
  int*   tok_e      = (int*)(ws + off);  off += (size_t)TDIM * TOPK * sizeof(int);
  float* tok_w      = (float*)(ws + off); off += (size_t)TDIM * TOPK * sizeof(float);
  int*   counts     = (int*)(ws + off);  off += EXP * sizeof(int);
  int*   offsets    = (int*)(ws + off);  off += EXP * sizeof(int);
  int*   cursors    = (int*)(ws + off);  off += EXP * sizeof(int);

  k_zero<<<2048, 256, 0, stream>>>(out, counts, (long)TDIM * HDIM);
  k_router<<<TDIM / 256, 256, 0, stream>>>(hidden, gate_w, tok_e, tok_w, counts);
  k_scan<<<1, 32, 0, stream>>>(counts, offsets, cursors);
  k_assign<<<TDIM / 256, 256, 0, stream>>>(tok_e, tok_w, cursors, row_token, row_weight);
  k_gemm1<<<dim3(TDIM / 64, IDIM / 128, EXP), 256, 0, stream>>>(
      hidden, w1, w3, row_token, offsets, counts, act);
  k_gemm2<<<dim3(TDIM / 64, HDIM / 128, EXP), 256, 0, stream>>>(
      act, w2, row_token, row_weight, offsets, counts, out);
}
